// ContinuousModel_58033598104185
// MI455X (gfx1250) — compile-verified, matrix operand order checked by reading, not run on hardware
//
#include <hip/hip_runtime.h>
#include <hip/hip_bf16.h>

#define N_NODES 10000
#define E_EDGES 2000000
#define MEM 64
#define MSG 64
#define RAW 128
#define GATE 192

typedef __attribute__((ext_vector_type(16))) __bf16 v16bf;
typedef __attribute__((ext_vector_type(8)))  float  v8f;

__device__ __forceinline__ unsigned short f2bf(float f) {
    unsigned int u = __float_as_uint(f);
    unsigned int r = u + 0x7FFFu + ((u >> 16) & 1u);   // round-to-nearest-even
    return (unsigned short)(r >> 16);
}

// ---------------- kernel 0: zero agg / cnt / tsmax ----------------
__global__ void k_zero(int* p, int n) {
    int i = blockIdx.x * blockDim.x + threadIdx.x;
    if (i < n) p[i] = 0;
}

// ---------------- kernel 1: edge scatter (segment sum / max) ----------------
// 64 lanes per edge: lane f handles feature f.
__global__ void k_edges(const int* __restrict__ src, const int* __restrict__ dst,
                        const float* __restrict__ td, const float* __restrict__ ts,
                        const float* __restrict__ memory,
                        float* __restrict__ agg, int* __restrict__ cnt,
                        int* __restrict__ tsmax) {
    int g = blockIdx.x * blockDim.x + threadIdx.x;
    int f = g & 63;
    int e0 = g >> 6;
    int estep = (gridDim.x * blockDim.x) >> 6;
    for (int e = e0; e < E_EDGES; e += estep) {
        int s = src[e];
        int d = dst[e];
        float scale = __expf(td[e] * (1.0f / 30.0f));
        float v = memory[d * MEM + f];
        atomicAdd(&agg[s * RAW + f], v);
        atomicAdd(&agg[s * RAW + MEM + f], v * scale);
        if (f == 0) {
            atomicAdd(&cnt[s], 1);
            atomicMax(&tsmax[s], __float_as_int(ts[e]));  // ts >= 0
        }
    }
}

// ---------------- kernel 2: per-node MLP + GRU + time-emb + Wp1 layer ----------------
// one block of 64 threads per node; emits emb2 in bf16 for the WMMA GEMM.
__global__ void k_node(const float* __restrict__ agg, const int* __restrict__ cnt,
                       const int* __restrict__ tsmax,
                       const float* __restrict__ memory, const float* __restrict__ last_update,
                       const int* __restrict__ now_time_p,
                       const float* __restrict__ W1, const float* __restrict__ b1,
                       const float* __restrict__ W2, const float* __restrict__ b2,
                       const float* __restrict__ W_ih, const float* __restrict__ W_hh,
                       const float* __restrict__ b_ih, const float* __restrict__ b_hh,
                       const float* __restrict__ Wt, const float* __restrict__ bt,
                       const float* __restrict__ Wp1, const float* __restrict__ bp1,
                       unsigned short* __restrict__ emb2) {
    __shared__ float sh_agg[RAW];
    __shared__ float sh_hid[MEM];
    __shared__ float sh_msg[MSG];
    __shared__ float sh_h[MEM];
    __shared__ float sh_emb[MEM];

    int node = blockIdx.x;
    int j = threadIdx.x;

    sh_agg[j]       = agg[node * RAW + j];
    sh_agg[j + 64]  = agg[node * RAW + j + 64];
    float h = memory[node * MEM + j];
    sh_h[j] = h;
    __syncthreads();

    // hid = relu(agg @ W1 + b1)   W1: [128,64]
    float s = b1[j];
    #pragma unroll 4
    for (int k = 0; k < RAW; ++k) s += sh_agg[k] * W1[k * 64 + j];
    sh_hid[j] = fmaxf(s, 0.0f);
    __syncthreads();

    // msg = hid @ W2 + b2         W2: [64,64]
    s = b2[j];
    #pragma unroll 4
    for (int k = 0; k < MEM; ++k) s += sh_hid[k] * W2[k * 64 + j];
    sh_msg[j] = s;
    __syncthreads();

    // GRU gates (r,z,n)
    float ir = b_ih[j], iz = b_ih[64 + j], in_ = b_ih[128 + j];
    float hr = b_hh[j], hz = b_hh[64 + j], hn = b_hh[128 + j];
    #pragma unroll 4
    for (int k = 0; k < MSG; ++k) {
        float m = sh_msg[k];
        ir += m * W_ih[k * GATE + j];
        iz += m * W_ih[k * GATE + 64 + j];
        in_ += m * W_ih[k * GATE + 128 + j];
    }
    #pragma unroll 4
    for (int k = 0; k < MEM; ++k) {
        float hv = sh_h[k];
        hr += hv * W_hh[k * GATE + j];
        hz += hv * W_hh[k * GATE + 64 + j];
        hn += hv * W_hh[k * GATE + 128 + j];
    }
    float r = 1.0f / (1.0f + __expf(-(ir + hr)));
    float z = 1.0f / (1.0f + __expf(-(iz + hz)));
    float n = tanhf(in_ + r * hn);
    float hnew = (1.0f - z) * n + z * h;

    bool has = cnt[node] > 0;
    float mem_upd = has ? hnew : h;
    float lastu = has ? __int_as_float(tsmax[node]) : last_update[node];
    float now = (float)(*now_time_p);
    float tdiff = now - lastu;

    // time embedding
    float emb = mem_upd * (1.0f + tdiff * Wt[j] + bt[j]);
    sh_emb[j] = emb;
    __syncthreads();

    // emb2 = relu(emb @ Wp1 + bp1), stored bf16
    s = bp1[j];
    #pragma unroll 4
    for (int k = 0; k < MEM; ++k) s += sh_emb[k] * Wp1[k * 64 + j];
    emb2[node * MEM + j] = f2bf(fmaxf(s, 0.0f));
}

// ---------------- kernel 2b: Wp2 [64,10000] f32 -> Wp2^T [10000,64] bf16 ----------------
__global__ void k_wp2t(const float* __restrict__ Wp2, unsigned short* __restrict__ wp2t) {
    int i = blockIdx.x * blockDim.x + threadIdx.x;   // i = n*64 + k
    if (i >= N_NODES * MEM) return;
    int k = i & 63;
    int n = i >> 6;
    wp2t[i] = f2bf(Wp2[k * N_NODES + n]);
}

// ---------------- kernel 3: od = emb2 @ Wp2 + bp2 via bf16 WMMA ----------------
// 16x16 tile per wave, K=64 -> 2x v_wmma_f32_16x16x32_bf16.
// Fragment layout (ISA 16-bit A 16x32): lane<16 -> row m, K{0..7,16..23};
// lane>=16 -> row m, K{8..15,24..31}. B columns use the symmetric layout via Wp2^T.
union FragU {
    uint4 u[2];
    v16bf v;
};

__global__ void k_gemm(const unsigned short* __restrict__ emb2,
                       const unsigned short* __restrict__ wp2t,
                       const float* __restrict__ bp2,
                       float* __restrict__ out) {
    int wave = threadIdx.x >> 5;
    int lane = threadIdx.x & 31;
    int half = lane >> 4;          // 0 or 1
    int lm = lane & 15;

    int tile_m = blockIdx.y;                 // 0..624
    int tile_n = blockIdx.x * 8 + wave;      // 8 waves/block across N
    if (tile_n >= 625) return;               // uniform per wave

    const unsigned short* ap = emb2 + (size_t)(tile_m * 16 + lm) * MEM + half * 8;
    const unsigned short* bp = wp2t + (size_t)(tile_n * 16 + lm) * MEM + half * 8;

    FragU a0, a1, b0, b1;
    a0.u[0] = *(const uint4*)(ap);        // K 0..7   (or 8..15)
    a0.u[1] = *(const uint4*)(ap + 16);   // K 16..23 (or 24..31)
    a1.u[0] = *(const uint4*)(ap + 32);   // K 32..39 (or 40..47)
    a1.u[1] = *(const uint4*)(ap + 48);   // K 48..55 (or 56..63)
    b0.u[0] = *(const uint4*)(bp);
    b0.u[1] = *(const uint4*)(bp + 16);
    b1.u[0] = *(const uint4*)(bp + 32);
    b1.u[1] = *(const uint4*)(bp + 48);

    float bias = bp2[tile_n * 16 + lm];
    v8f c = {bias, bias, bias, bias, bias, bias, bias, bias};

    c = __builtin_amdgcn_wmma_f32_16x16x32_bf16(false, a0.v, false, b0.v,
                                                (short)0, c, false, false);
    c = __builtin_amdgcn_wmma_f32_16x16x32_bf16(false, a1.v, false, b1.v,
                                                (short)0, c, false, false);

    // C/D layout: VGPR r -> (M = r + 8*half, N = lm)
    int col = tile_n * 16 + lm;
    #pragma unroll
    for (int r = 0; r < 8; ++r) {
        int row = tile_m * 16 + r + half * 8;
        out[(size_t)row * N_NODES + col] = c[r];
    }
}

extern "C" void kernel_launch(void* const* d_in, const int* in_sizes, int n_in,
                              void* d_out, int out_size, void* d_ws, size_t ws_size,
                              hipStream_t stream) {
    const int*   src     = (const int*)d_in[0];
    const int*   dst     = (const int*)d_in[1];
    const float* td      = (const float*)d_in[2];
    const float* ts      = (const float*)d_in[3];
    const int*   now_p   = (const int*)d_in[4];
    const float* memory  = (const float*)d_in[5];
    const float* last_up = (const float*)d_in[6];
    const float* W1      = (const float*)d_in[7];
    const float* b1      = (const float*)d_in[8];
    const float* W2      = (const float*)d_in[9];
    const float* b2      = (const float*)d_in[10];
    const float* W_ih    = (const float*)d_in[11];
    const float* W_hh    = (const float*)d_in[12];
    const float* b_ih    = (const float*)d_in[13];
    const float* b_hh    = (const float*)d_in[14];
    const float* Wt      = (const float*)d_in[15];
    const float* bt      = (const float*)d_in[16];
    const float* Wp1     = (const float*)d_in[17];
    const float* bp1     = (const float*)d_in[18];
    const float* Wp2     = (const float*)d_in[19];
    const float* bp2     = (const float*)d_in[20];
    float* out = (float*)d_out;

    // workspace carve-up
    char* ws = (char*)d_ws;
    float* agg   = (float*)ws;                                    // N*128 f32
    int*   cnt   = (int*)(ws + (size_t)N_NODES * RAW * 4);        // N
    int*   tsmax = cnt + N_NODES;                                 // N
    unsigned short* emb2 = (unsigned short*)(tsmax + N_NODES);    // N*64 bf16
    unsigned short* wp2t = emb2 + (size_t)N_NODES * MEM;          // 10000*64 bf16

    // 0) zero agg + cnt + tsmax (N*130 dwords)
    int nz = N_NODES * (RAW + 2);
    k_zero<<<(nz + 255) / 256, 256, 0, stream>>>((int*)ws, nz);

    // 1) edge scatter
    k_edges<<<2048, 256, 0, stream>>>(src, dst, td, ts, memory, agg, cnt, tsmax);

    // 2b) Wp2 transpose+bf16 (independent of node kernel)
    k_wp2t<<<(N_NODES * MEM + 255) / 256, 256, 0, stream>>>(Wp2, wp2t);

    // 2) per-node update -> emb2 (bf16)
    k_node<<<N_NODES, 64, 0, stream>>>(agg, cnt, tsmax, memory, last_up, now_p,
                                       W1, b1, W2, b2, W_ih, W_hh, b_ih, b_hh,
                                       Wt, bt, Wp1, bp1, emb2);

    // 3) od = emb2 @ Wp2 + bp2   (625x625 WMMA tiles, 8 tiles/block)
    dim3 grid((625 + 7) / 8, 625);
    k_gemm<<<grid, 256, 0, stream>>>(emb2, wp2t, bp2, out);
}
